// UNet_44581760532913
// MI455X (gfx1250) — matrix operand order repaired and checked by reading, not compiled, not run on hardware
//
#include <hip/hip_runtime.h>

typedef __attribute__((ext_vector_type(16))) _Float16 v16h;
typedef __attribute__((ext_vector_type(8)))  _Float16 v8h;
typedef __attribute__((ext_vector_type(8)))  float    v8f;

// ------------------------- constants -------------------------
static constexpr int HLOG  = 21;
static constexpr int HCAP  = 1 << HLOG;            // 2M-slot open-addressing hash
static constexpr unsigned HMASK = HCAP - 1;
static constexpr unsigned long long HEMPTY = ~0ull;
static constexpr int N1CAP = 300000;               // level-1 voxel capacity (<=65^3 cells)
static constexpr int N2CAP = 60000;                // level-2 voxel capacity

// ------------------------- device helpers -------------------------
__device__ __forceinline__ unsigned long long enc3(int x, int y, int z) {
  return (((unsigned long long)(unsigned)(x + 64)) * 4096ull
          + (unsigned)(y + 64)) * 4096ull + (unsigned)(z + 64);
}
__device__ __forceinline__ unsigned hslot(unsigned long long k) {
  return (unsigned)((k * 0x9E3779B97F4A7C15ull) >> 40) & HMASK;
}
__device__ __forceinline__ int hlookup(const unsigned long long* K, const int* V,
                                       unsigned long long key) {
  unsigned h = hslot(key);
  for (int t = 0; t < HCAP; ++t) {
    unsigned long long cur = K[h];
    if (cur == key) return V[h];
    if (cur == HEMPTY) return -1;
    h = (h + 1) & HMASK;
  }
  return -1;
}

// ------------------------- tiny utility kernels -------------------------
__global__ void k_zero_f32(float* p, long n) {
  long i = (long)blockIdx.x * blockDim.x + threadIdx.x;
  if (i < n) p[i] = 0.0f;
}
__global__ void k_zero_i32(int* p, long n) {
  long i = (long)blockIdx.x * blockDim.x + threadIdx.x;
  if (i < n) p[i] = 0;
}
__global__ void k_set_i32(int* p, int v) { *p = v; }
__global__ void k_hash_clear(unsigned long long* K) {
  long i = (long)blockIdx.x * blockDim.x + threadIdx.x;
  if (i < HCAP) K[i] = HEMPTY;
}

// repack f32 weights [k][CIN][COUT] -> f16 B-operand layout [k][CIN/32][COUT][32]
__global__ void k_pack_w(const float* __restrict__ W, _Float16* __restrict__ Wh,
                         long total, int CIN, int COUT) {
  long i = (long)blockIdx.x * blockDim.x + threadIdx.x;
  if (i >= total) return;
  int cout = (int)(i % COUT);
  long r = i / COUT;
  int cin = (int)(r % CIN);
  int k = (int)(r / CIN);
  Wh[(((size_t)k * (CIN / 32) + (cin >> 5)) * COUT + cout) * 32 + (cin & 31)] =
      (_Float16)W[i];
}

// ------------------------- structure build -------------------------
__global__ void k_hash_build(const int* __restrict__ cells, const int* nptr, int ncap,
                             unsigned long long* K, int* V) {
  int n = *nptr; if (n > ncap) n = ncap;
  int i = blockIdx.x * blockDim.x + threadIdx.x;
  if (i >= n) return;
  unsigned long long key = enc3(cells[3 * i], cells[3 * i + 1], cells[3 * i + 2]);
  unsigned h = hslot(key);
  for (int t = 0; t < HCAP; ++t) {
    unsigned long long old = atomicCAS(&K[h], HEMPTY, key);
    if (old == HEMPTY) { V[h] = i; return; }
    if (old == key) return;               // duplicate (shouldn't happen)
    h = (h + 1) & HMASK;
  }
}

// submanifold K=3: neighbor cells c+d, d in {-1,0,1}^3 ; k = ((dx+1)*3+(dy+1))*3+(dz+1)
__global__ void k_build_sm_pairs(const int* __restrict__ cells, const int* nptr, int ncap,
                                 const unsigned long long* __restrict__ K,
                                 const int* __restrict__ V,
                                 int* pin, int* pout, int* cnt, int pairCap) {
  int n = *nptr; if (n > ncap) n = ncap;
  int i = blockIdx.x * blockDim.x + threadIdx.x;
  if (i >= n) return;
  int cx = cells[3 * i], cy = cells[3 * i + 1], cz = cells[3 * i + 2];
  int kk = 0;
  for (int dx = -1; dx <= 1; ++dx)
    for (int dy = -1; dy <= 1; ++dy)
      for (int dz = -1; dz <= 1; ++dz) {
        int j = hlookup(K, V, enc3(cx + dx, cy + dy, cz + dz));
        if (j >= 0) {
          int s = atomicAdd(&cnt[kk], 1);
          if (s < pairCap) {
            pin[(size_t)kk * pairCap + s] = j;    // input (gather) index
            pout[(size_t)kk * pairCap + s] = i;   // output (scatter) index
          }
        }
        ++kk;
      }
}

// downsample dedup: out cells = unique even e = c + f, f in {-1,0}^3, all >= 0
__global__ void k_build_down_dedup(const int* __restrict__ cells, const int* nptr, int ncap,
                                   unsigned long long* K, int* outCells, int* outCnt,
                                   int outCap) {
  int n = *nptr; if (n > ncap) n = ncap;
  int i = blockIdx.x * blockDim.x + threadIdx.x;
  if (i >= n) return;
  int cx = cells[3 * i], cy = cells[3 * i + 1], cz = cells[3 * i + 2];
  for (int b = 0; b < 8; ++b) {
    int ex = cx - ((b >> 2) & 1), ey = cy - ((b >> 1) & 1), ez = cz - (b & 1);
    if (ex < 0 || ey < 0 || ez < 0) continue;
    if ((ex | ey | ez) & 1) continue;     // must be even in every dim
    unsigned long long key = enc3(ex, ey, ez);
    unsigned h = hslot(key);
    for (int t = 0; t < HCAP; ++t) {
      unsigned long long old = atomicCAS(&K[h], HEMPTY, key);
      if (old == HEMPTY) {
        int idx = atomicAdd(outCnt, 1);
        if (idx < outCap) {
          outCells[3 * idx] = ex; outCells[3 * idx + 1] = ey; outCells[3 * idx + 2] = ez;
        }
        break;
      }
      if (old == key) break;
      h = (h + 1) & HMASK;
    }
  }
}

// strided K=2 pairs: input cell = e + idx, idx in {0,1}^3 ; k = i0*4+i1*2+i2.
// Afterwards converts stored cell e -> next-level cell e>>1.
__global__ void k_build_down_pairs(int* __restrict__ eCells, const int* nptr, int ncap,
                                   const unsigned long long* __restrict__ K,
                                   const int* __restrict__ V,
                                   int* pin, int* pout, int* cnt, int pairCap) {
  int n = *nptr; if (n > ncap) n = ncap;
  int o = blockIdx.x * blockDim.x + threadIdx.x;
  if (o >= n) return;
  int ex = eCells[3 * o], ey = eCells[3 * o + 1], ez = eCells[3 * o + 2];
  for (int b = 0; b < 8; ++b) {
    int j = hlookup(K, V, enc3(ex + ((b >> 2) & 1), ey + ((b >> 1) & 1), ez + (b & 1)));
    if (j >= 0) {
      int s = atomicAdd(&cnt[b], 1);
      if (s < pairCap) {
        pin[(size_t)b * pairCap + s] = j;
        pout[(size_t)b * pairCap + s] = o;
      }
    }
  }
  eCells[3 * o] = ex >> 1; eCells[3 * o + 1] = ey >> 1; eCells[3 * o + 2] = ez >> 1;
}

// ------------------------- batchnorm(+ReLU) -------------------------
__global__ void k_bn_stats(const float* __restrict__ x, const int* nptr, int ncap, int C,
                           float* sums) {
  int n = *nptr; if (n > ncap) n = ncap;
  int g = blockIdx.x * blockDim.x + threadIdx.x;   // grid = C blocks of 256
  int c = g % C, r0 = g / C;
  int stride = (gridDim.x * blockDim.x) / C;
  float s = 0.f, q = 0.f;
  for (int r = r0; r < n; r += stride) {
    float v = x[(size_t)r * C + c];
    s += v; q += v * v;
  }
  atomicAdd(&sums[c], s);
  atomicAdd(&sums[C + c], q);
}

// BN + ReLU, emitting f16 activations (conv inputs live in f16 at rest).
__global__ void k_bn_apply(const float* __restrict__ x, _Float16* __restrict__ y,
                           const int* nptr, int ncap, int C, const float* __restrict__ sums,
                           const float* __restrict__ w, const float* __restrict__ b) {
  int n = *nptr; if (n > ncap) n = ncap;
  long i = (long)blockIdx.x * blockDim.x + threadIdx.x;
  if (i >= (long)n * C) return;
  int c = (int)(i % C);
  float inv = 1.0f / (float)n;
  float m = sums[c] * inv;
  float v = sums[C + c] * inv - m * m;
  float r = rsqrtf(v + 1e-4f);
  float val = (x[i] - m) * r * w[c] + b[c];
  y[i] = (_Float16)(val > 0.f ? val : 0.f);
}

// copy skip features into left columns of a wider (concat) buffer
__global__ void k_copy_cols(const float* __restrict__ src, int srcC,
                            float* __restrict__ dst, int dstC, int colOff,
                            const int* nptr, int ncap) {
  int n = *nptr; if (n > ncap) n = ncap;
  long i = (long)blockIdx.x * blockDim.x + threadIdx.x;
  if (i >= (long)n * srcC) return;
  long r = i / srcC; int c = (int)(i % srcC);
  dst[r * dstC + colOff + c] = src[i];
}

// ------------------------- WMMA gather-GEMM-scatter -------------------------
// One wave handles 16 rulebook pairs. A = gathered f16 feature rows (16 x CIN),
// loaded as 2x16B vectors per K-chunk; B = pre-packed f16 weights, one aligned
// 32B v16h load per lane; C/D = f32 accumulators; scatter via global_atomic_add_f32.
// D layout per CDNA5 ISA 7.12.2 (wave32, v_wmma_f32_16x16x32_f16); A/B use contiguous k = 16*half + s.
template <int CIN, int COUT>
__global__ __launch_bounds__(256)
void spconv_wmma(const _Float16* __restrict__ feat,
                 const _Float16* __restrict__ Wh,  // [k][CIN/32][COUT][32] f16
                 const int* __restrict__ gidx,     // gather rows, [Kelems*pairCap]
                 const int* __restrict__ sidx,     // scatter rows
                 const int* __restrict__ cnt,      // per-k pair counts
                 int pairCap,
                 float* __restrict__ out, int outStride, int colOff) {
  const int k    = blockIdx.y;
  const int lane = threadIdx.x & 31;
  const int wv   = threadIdx.x >> 5;
  const int tile = blockIdx.x * (blockDim.x >> 5) + wv;
  int npairs = cnt[k]; if (npairs > pairCap) npairs = pairCap;
  const int p0 = tile * 16;
  if (p0 >= npairs) return;                        // wave-uniform exit (EXEC stays full)

  const int half = lane >> 4;                      // 0: lanes 0-15, 1: lanes 16-31
  const int m    = lane & 15;
  const int* gk = gidx + (size_t)k * pairCap;
  const int* sk = sidx + (size_t)k * pairCap;
  const _Float16* Wk = Wh + (size_t)k * (CIN / 32) * COUT * 32;

  const int pa = p0 + m;
  const int grow = (pa < npairs) ? gk[pa] : -1;
  const bool aok = grow >= 0;
  const _Float16* arow = feat + (size_t)(aok ? grow : 0) * CIN;

  v8f zero;
#pragma unroll
  for (int j = 0; j < 8; ++j) zero[j] = 0.f;
  v8f acc[COUT / 16];
#pragma unroll
  for (int t = 0; t < COUT / 16; ++t) acc[t] = zero;

#pragma unroll
  for (int kb = 0; kb < CIN; kb += 32) {
    // A 16x32 f16: lane half0 holds K kb+0..15, half1 kb+16..31
    v16h a;
    const int ko = half * 16;
    if (aok) {
      v8h lo = *(const v8h*)(arow + kb + ko);
      v8h hi = *(const v8h*)(arow + kb + 8 + ko);
#pragma unroll
      for (int e = 0; e < 8; ++e) { a[e] = lo[e]; a[e + 8] = hi[e]; }
    } else {
#pragma unroll
      for (int e = 0; e < 16; ++e) a[e] = (_Float16)0.f;
    }
#pragma unroll
    for (int t = 0; t < COUT / 16; ++t) {
      // B 32x16 f16: col = t*16+m, half0 holds K kb+0..15, half1 kb+16..31
      const _Float16* wp =
          Wk + (((size_t)(kb >> 5) * COUT) + t * 16 + m) * 32 + half * 16;
      v16h b = *(const v16h*)wp;                    // 32B aligned
      acc[t] = __builtin_amdgcn_wmma_f32_16x16x32_f16(false, a, false, b,
                                                      (short)0, acc[t], false, false);
    }
  }

  // D 16x16 f32: VGPR j -> row j + half*8, col m
#pragma unroll
  for (int t = 0; t < COUT / 16; ++t) {
#pragma unroll
    for (int j = 0; j < 8; ++j) {
      int p = p0 + j + half * 8;
      if (p < npairs) {
        int sr = sk[p];
        atomicAdd(out + (size_t)sr * outStride + colOff + t * 16 + m, acc[t][j]);
      }
    }
  }
}

// ------------------------- host orchestration -------------------------
extern "C" void kernel_launch(void* const* d_in, const int* in_sizes, int n_in,
                              void* d_out, int out_size, void* d_ws, size_t ws_size,
                              hipStream_t stream) {
  (void)n_in; (void)out_size; (void)ws_size;
  // Inputs flattened pytree-style (sorted keys): feat, 27 params sorted, pos.
  const float* feat = (const float*)d_in[0];
  const float* bn0_b = (const float*)d_in[1],  *bn0_w = (const float*)d_in[2];
  const float* bn1_b = (const float*)d_in[3],  *bn1_w = (const float*)d_in[4];
  const float* bn2_b = (const float*)d_in[5],  *bn2_w = (const float*)d_in[6];
  const float* bnd0_b = (const float*)d_in[7], *bnd0_w = (const float*)d_in[8];
  const float* bnd1_b = (const float*)d_in[9], *bnd1_w = (const float*)d_in[10];
  const float* bnp0_b = (const float*)d_in[11],*bnp0_w = (const float*)d_in[12];
  const float* bnp1_b = (const float*)d_in[13],*bnp1_w = (const float*)d_in[14];
  const float* bnu0_b = (const float*)d_in[15],*bnu0_w = (const float*)d_in[16];
  const float* bnu1_b = (const float*)d_in[17],*bnu1_w = (const float*)d_in[18];
  const float* cd0W  = (const float*)d_in[19], *cd1W  = (const float*)d_in[20];
  const float* du0W  = (const float*)d_in[21], *du1W  = (const float*)d_in[22];
  const float* sm0W  = (const float*)d_in[23], *sm1W  = (const float*)d_in[24];
  const float* sm2W  = (const float*)d_in[25];
  const float* smp0W = (const float*)d_in[26], *smp1W = (const float*)d_in[27];
  const int*   pos   = (const int*)d_in[28];
  const int N0 = in_sizes[28] / 3;

  // -------- workspace carve-out --------
  char* base = (char*)d_ws;
  size_t off = 0;
  auto alloc = [&](size_t bytes) -> void* {
    void* p = base + off; off = (off + bytes + 255) & ~(size_t)255; return p;
  };
  const size_t MAXT = (size_t)N0 * 64 > (size_t)N1CAP * 128 ? (size_t)N0 * 64
                                                            : (size_t)N1CAP * 128;
  float* S0 = (float*)alloc((size_t)N0 * 32 * 4);       // skip0
  float* S1 = (float*)alloc((size_t)N1CAP * 64 * 4);    // skip1
  _Float16* T1h = (_Float16*)alloc(MAXT * 2);           // f16 activations (conv inputs)
  float* T2 = (float*)alloc(MAXT * 4);
  float* T3 = (float*)alloc((size_t)N1CAP * 64 * 4);
  float* bnsums = (float*)alloc(512 * 4);
  int* cells1 = (int*)alloc((size_t)N1CAP * 3 * 4);
  int* cells2 = (int*)alloc((size_t)N2CAP * 3 * 4);
  int* pin0  = (int*)alloc((size_t)27 * N0 * 4);
  int* pout0 = (int*)alloc((size_t)27 * N0 * 4);
  int* pin1  = (int*)alloc((size_t)27 * N1CAP * 4);
  int* pout1 = (int*)alloc((size_t)27 * N1CAP * 4);
  int* pin2  = (int*)alloc((size_t)27 * N2CAP * 4);
  int* pout2 = (int*)alloc((size_t)27 * N2CAP * 4);
  int* dpin0  = (int*)alloc((size_t)8 * N1CAP * 4);
  int* dpout0 = (int*)alloc((size_t)8 * N1CAP * 4);
  int* dpin1  = (int*)alloc((size_t)8 * N2CAP * 4);
  int* dpout1 = (int*)alloc((size_t)8 * N2CAP * 4);
  int* cntSm0 = (int*)alloc(27 * 4);
  int* cntSm1 = (int*)alloc(27 * 4);
  int* cntSm2 = (int*)alloc(27 * 4);
  int* cntDn0 = (int*)alloc(8 * 4);
  int* cntDn1 = (int*)alloc(8 * 4);
  int* ndev   = (int*)alloc(16);                        // n0, n1, n2
  unsigned long long* hkA = (unsigned long long*)alloc((size_t)HCAP * 8);
  int* hvA = (int*)alloc((size_t)HCAP * 4);
  unsigned long long* hkB = (unsigned long long*)alloc((size_t)HCAP * 8);
  // packed f16 weights
  _Float16* sm0H  = (_Float16*)alloc((size_t)27 * 32 * 32 * 2);
  _Float16* cd0H  = (_Float16*)alloc((size_t)8 * 32 * 64 * 2);
  _Float16* sm1H  = (_Float16*)alloc((size_t)27 * 64 * 64 * 2);
  _Float16* cd1H  = (_Float16*)alloc((size_t)8 * 64 * 96 * 2);
  _Float16* sm2H  = (_Float16*)alloc((size_t)27 * 96 * 96 * 2);
  _Float16* du1H  = (_Float16*)alloc((size_t)8 * 96 * 64 * 2);
  _Float16* smp1H = (_Float16*)alloc((size_t)27 * 128 * 64 * 2);
  _Float16* du0H  = (_Float16*)alloc((size_t)8 * 64 * 32 * 2);
  _Float16* smp0H = (_Float16*)alloc((size_t)27 * 64 * 32 * 2);

  auto blks = [](long n) { return (unsigned)((n + 255) / 256); };
  auto zf = [&](float* p, long n) { k_zero_f32<<<blks(n), 256, 0, stream>>>(p, n); };
  auto zi = [&](int* p, long n)   { k_zero_i32<<<blks(n), 256, 0, stream>>>(p, n); };
  auto bn = [&](const float* x, _Float16* y, const int* nptr, int ncap, int C,
                const float* w, const float* b) {
    zf(bnsums, 512);
    k_bn_stats<<<C, 256, 0, stream>>>(x, nptr, ncap, C, bnsums);
    k_bn_apply<<<blks((long)ncap * C), 256, 0, stream>>>(x, y, nptr, ncap, C, bnsums, w, b);
  };
  auto pack = [&](const float* W, _Float16* Wh, int Kel, int CIN, int COUT) {
    long tot = (long)Kel * CIN * COUT;
    k_pack_w<<<blks(tot), 256, 0, stream>>>(W, Wh, tot, CIN, COUT);
  };

  // ================= Phase A: rulebook construction =================
  k_set_i32<<<1, 1, 0, stream>>>(&ndev[0], N0);
  // weight repacks (independent of structure build)
  pack(sm0W, sm0H, 27, 32, 32);   pack(cd0W, cd0H, 8, 32, 64);
  pack(sm1W, sm1H, 27, 64, 64);   pack(cd1W, cd1H, 8, 64, 96);
  pack(sm2W, sm2H, 27, 96, 96);   pack(du1W, du1H, 8, 96, 64);
  pack(smp1W, smp1H, 27, 128, 64);pack(du0W, du0H, 8, 64, 32);
  pack(smp0W, smp0H, 27, 64, 32);
  // level 0
  k_hash_clear<<<blks(HCAP), 256, 0, stream>>>(hkA);
  k_hash_build<<<blks(N0), 256, 0, stream>>>(pos, &ndev[0], N0, hkA, hvA);
  zi(cntSm0, 27);
  k_build_sm_pairs<<<blks(N0), 256, 0, stream>>>(pos, &ndev[0], N0, hkA, hvA,
                                                 pin0, pout0, cntSm0, N0);
  k_hash_clear<<<blks(HCAP), 256, 0, stream>>>(hkB);
  k_set_i32<<<1, 1, 0, stream>>>(&ndev[1], 0);
  k_build_down_dedup<<<blks(N0), 256, 0, stream>>>(pos, &ndev[0], N0, hkB,
                                                   cells1, &ndev[1], N1CAP);
  zi(cntDn0, 8);
  k_build_down_pairs<<<blks(N1CAP), 256, 0, stream>>>(cells1, &ndev[1], N1CAP, hkA, hvA,
                                                      dpin0, dpout0, cntDn0, N1CAP);
  // level 1
  k_hash_clear<<<blks(HCAP), 256, 0, stream>>>(hkA);
  k_hash_build<<<blks(N1CAP), 256, 0, stream>>>(cells1, &ndev[1], N1CAP, hkA, hvA);
  zi(cntSm1, 27);
  k_build_sm_pairs<<<blks(N1CAP), 256, 0, stream>>>(cells1, &ndev[1], N1CAP, hkA, hvA,
                                                    pin1, pout1, cntSm1, N1CAP);
  k_hash_clear<<<blks(HCAP), 256, 0, stream>>>(hkB);
  k_set_i32<<<1, 1, 0, stream>>>(&ndev[2], 0);
  k_build_down_dedup<<<blks(N1CAP), 256, 0, stream>>>(cells1, &ndev[1], N1CAP, hkB,
                                                      cells2, &ndev[2], N2CAP);
  zi(cntDn1, 8);
  k_build_down_pairs<<<blks(N2CAP), 256, 0, stream>>>(cells2, &ndev[2], N2CAP, hkA, hvA,
                                                      dpin1, dpout1, cntDn1, N2CAP);
  // level 2
  k_hash_clear<<<blks(HCAP), 256, 0, stream>>>(hkA);
  k_hash_build<<<blks(N2CAP), 256, 0, stream>>>(cells2, &ndev[2], N2CAP, hkA, hvA);
  zi(cntSm2, 27);
  k_build_sm_pairs<<<blks(N2CAP), 256, 0, stream>>>(cells2, &ndev[2], N2CAP, hkA, hvA,
                                                    pin2, pout2, cntSm2, N2CAP);

  // ================= Phase B: UNet compute =================
  dim3 blk(256);
  auto tg = [](int cap, int K) { return dim3((unsigned)((cap + 127) / 128), (unsigned)K); };

  // level 0 down path
  bn(feat, T1h, &ndev[0], N0, 32, bn0_w, bn0_b);
  zf(S0, (long)N0 * 32);
  spconv_wmma<32, 32><<<tg(N0, 27), blk, 0, stream>>>(T1h, sm0H, pin0, pout0, cntSm0, N0, S0, 32, 0);
  bn(S0, T1h, &ndev[0], N0, 32, bnd0_w, bnd0_b);
  zf(T2, (long)N1CAP * 64);
  spconv_wmma<32, 64><<<tg(N1CAP, 8), blk, 0, stream>>>(T1h, cd0H, dpin0, dpout0, cntDn0, N1CAP, T2, 64, 0);
  // level 1 down path
  bn(T2, T1h, &ndev[1], N1CAP, 64, bn1_w, bn1_b);
  zf(S1, (long)N1CAP * 64);
  spconv_wmma<64, 64><<<tg(N1CAP, 27), blk, 0, stream>>>(T1h, sm1H, pin1, pout1, cntSm1, N1CAP, S1, 64, 0);
  bn(S1, T1h, &ndev[1], N1CAP, 64, bnd1_w, bnd1_b);
  zf(T2, (long)N2CAP * 96);
  spconv_wmma<64, 96><<<tg(N2CAP, 8), blk, 0, stream>>>(T1h, cd1H, dpin1, dpout1, cntDn1, N2CAP, T2, 96, 0);
  // level 2 (bottom)
  bn(T2, T1h, &ndev[2], N2CAP, 96, bn2_w, bn2_b);
  zf(T3, (long)N2CAP * 96);
  spconv_wmma<96, 96><<<tg(N2CAP, 27), blk, 0, stream>>>(T1h, sm2H, pin2, pout2, cntSm2, N2CAP, T3, 96, 0);
  // up to level 1: transpose conv (gather=coarse pout, scatter=fine pin) + concat
  bn(T3, T1h, &ndev[2], N2CAP, 96, bnu1_w, bnu1_b);
  zf(T2, (long)N1CAP * 128);
  k_copy_cols<<<blks((long)N1CAP * 64), 256, 0, stream>>>(S1, 64, T2, 128, 0, &ndev[1], N1CAP);
  spconv_wmma<96, 64><<<tg(N2CAP, 8), blk, 0, stream>>>(T1h, du1H, dpout1, dpin1, cntDn1, N2CAP, T2, 128, 64);
  bn(T2, T1h, &ndev[1], N1CAP, 128, bnp1_w, bnp1_b);
  zf(T3, (long)N1CAP * 64);
  spconv_wmma<128, 64><<<tg(N1CAP, 27), blk, 0, stream>>>(T1h, smp1H, pin1, pout1, cntSm1, N1CAP, T3, 64, 0);
  // up to level 0: transpose conv + concat
  bn(T3, T1h, &ndev[1], N1CAP, 64, bnu0_w, bnu0_b);
  zf(T2, (long)N0 * 64);
  k_copy_cols<<<blks((long)N0 * 32), 256, 0, stream>>>(S0, 32, T2, 64, 0, &ndev[0], N0);
  spconv_wmma<64, 32><<<tg(N1CAP, 8), blk, 0, stream>>>(T1h, du0H, dpout0, dpin0, cntDn0, N1CAP, T2, 64, 32);
  bn(T2, T1h, &ndev[0], N0, 64, bnp0_w, bnp0_b);
  // final submanifold conv straight into d_out
  float* outp = (float*)d_out;
  zf(outp, (long)N0 * 32);
  spconv_wmma<64, 32><<<tg(N0, 27), blk, 0, stream>>>(T1h, smp0H, pin0, pout0, cntSm0, N0, outp, 32, 0);
}
